// VectorQuantize_47811575939676
// MI455X (gfx1250) — matrix operand order repaired and checked by reading, not compiled, not run on hardware
//
#include <hip/hip_runtime.h>
#include <hip/hip_bf16.h>

#define BN   32768          // B*N points
#define DIM  256            // feature dim
#define KC   8192           // codebook entries

typedef __attribute__((ext_vector_type(16))) __bf16 bf16x16;
typedef __attribute__((ext_vector_type(8)))  __bf16 bf16x8;
typedef __attribute__((ext_vector_type(8)))  float  f32x8;
typedef __attribute__((ext_vector_type(4)))  int    i32x4;

#if __has_builtin(__builtin_amdgcn_global_load_async_to_lds_b128)
#define HAS_ASYNC_LDS 1
#else
#define HAS_ASYNC_LDS 0
#endif

// builtin expects int4* in global (AS1) / shared (AS3) address spaces
typedef __attribute__((address_space(1))) i32x4* gv4_ptr;
typedef __attribute__((address_space(3))) i32x4* lv4_ptr;

// ---- workspace layout (bytes) ----
#define WS_XB   0u                    // bf16 x       : BN*DIM*2  = 16,777,216
#define WS_CB   16777216u             // bf16 codebook: KC*DIM*2  =  4,194,304
#define WS_X2   20971520u             // f32 ||x||^2  : BN*4      =    131,072
#define WS_C2   21102592u             // f32 ||c||^2  : KC*4      =     32,768
#define WS_IND  21135360u             // i32 argmin   : BN*4      =    131,072

// ---- output layout (floats), reference tuple order ----
#define OUT_Q    0u            // quantize_st [8,4096,256]
#define OUT_IND  8388608u      // embed_ind   [8,4096] (stored as f32)
#define OUT_LOSS 8421376u      // commit_loss scalar
#define OUT_CS   8421377u      // cluster_size[8192]
#define OUT_ES   8429569u      // embed_sum   [8192,256]

// -------------------------------------------------------------------
// 1) convert to bf16 + squared norms. One 256-thread block per row
//    of the combined (x rows ++ codebook rows) space.
// -------------------------------------------------------------------
__global__ void __launch_bounds__(256) vq_prep(
    const float* __restrict__ x, const float* __restrict__ cb,
    __bf16* __restrict__ xb, __bf16* __restrict__ cbb,
    float* __restrict__ x2, float* __restrict__ c2)
{
    __shared__ float red[256];
    const int row = blockIdx.x;
    const int t   = threadIdx.x;
    const float* src; __bf16* dst; float* sq; int r;
    if (row < BN) { r = row;      src = x  + (size_t)r * DIM; dst = xb  + (size_t)r * DIM; sq = x2; }
    else          { r = row - BN; src = cb + (size_t)r * DIM; dst = cbb + (size_t)r * DIM; sq = c2; }
    float v = src[t];
    dst[t]  = (__bf16)v;
    red[t]  = v * v;
    __syncthreads();
    for (int s = 128; s > 0; s >>= 1) { if (t < s) red[t] += red[t + s]; __syncthreads(); }
    if (t == 0) sq[r] = red[0];
}

// -------------------------------------------------------------------
// 2) fused GEMM (x @ codebook^T via bf16 WMMA, f32 accum) + row argmin
//    128 threads = 4 waves; wave w owns 32 rows (TWO 16-row A sets,
//    both register-resident for the whole K sweep) so each B fragment
//    read from LDS feeds TWO WMMAs -> halves LDS bandwidth per FLOP.
//    Codebook tiles (32 rows) are double-buffered in LDS via async
//    global->LDS DMA: prefetch tile i+1 while computing tile i.
// -------------------------------------------------------------------
#define LDSD  272   // 256 + 16 halves row pitch: breaks LDS bank alignment
#define TROWS 32    // codebook rows per LDS tile

__global__ void __launch_bounds__(128) vq_argmin(
    const __bf16* __restrict__ xb, const __bf16* __restrict__ cbb,
    const float* __restrict__ x2, const float* __restrict__ c2,
    int* __restrict__ ind_ws, float* __restrict__ ind_out)
{
    __shared__ __bf16 lds[2][TROWS * LDSD];
    const int t    = threadIdx.x;
    const int wave = t >> 5;
    const int lane = t & 31;
    const int hl   = lane >> 4;    // half-wave selector
    const int l15  = lane & 15;

    const int m_base = blockIdx.x * 128 + wave * 32;   // 32 rows per wave

    // cooperative-stage mapping: 128 threads cover 32 rows x 256 halves
    const int r_ld = t >> 2;       // 0..31 tile row
    const int seg  = t & 3;        // 4 segments x 64 halves (128 B = 8 x b128)
    const __bf16* gsrc0 = cbb + (size_t)r_ld * DIM + seg * 64;

    // A fragments for two 16-row M sets; ISA 16-bit A layout:
    //   lanes 0-15 hold K = [0..7] & [16..23]; lanes 16-31 hold [8..15] & [24..31]
    bf16x16 afrag[2][8];
    #pragma unroll
    for (int s = 0; s < 2; ++s) {
        const __bf16* arow = xb + (size_t)(m_base + s * 16 + l15) * DIM;
        #pragma unroll
        for (int c = 0; c < 8; ++c) {
            bf16x8 lo = *(const bf16x8*)(arow + c * 32 + hl * 8);
            bf16x8 hi = *(const bf16x8*)(arow + c * 32 + 16 + hl * 8);
            afrag[s][c] = __builtin_shufflevector(lo, hi, 0,1,2,3,4,5,6,7,8,9,10,11,12,13,14,15);
        }
    }

    // C-fragment rows for this lane: m_base + s*16 + hl*8 + r (r = vgpr index)
    float x2row[2][8];
    #pragma unroll
    for (int s = 0; s < 2; ++s)
        #pragma unroll
        for (int r = 0; r < 8; ++r) x2row[s][r] = x2[m_base + s * 16 + hl * 8 + r];

    float bestd[2][8]; int besti[2][8];
    #pragma unroll
    for (int s = 0; s < 2; ++s)
        #pragma unroll
        for (int r = 0; r < 8; ++r) { bestd[s][r] = 3.4e38f; besti[s][r] = 0; }

    // ---- tile prefetch: this thread's share of 32 codebook rows -> LDS ----
    auto prefetch = [&](int buf, int n0) {
        const __bf16* src = gsrc0 + (size_t)n0 * DIM;
        __bf16*       dst = &lds[buf][r_ld * LDSD + seg * 64];
#if HAS_ASYNC_LDS
        #pragma unroll
        for (int i = 0; i < 8; ++i)
            __builtin_amdgcn_global_load_async_to_lds_b128(
                (gv4_ptr)(uintptr_t)(src + i * 8),
                (lv4_ptr)(uint32_t)(uintptr_t)(dst + i * 8), 0, 0);
#else
        const uint4* s4 = (const uint4*)src;
        uint4*       d4 = (uint4*)dst;
        #pragma unroll
        for (int i = 0; i < 8; ++i) d4[i] = s4[i];
#endif
    };

    prefetch(0, 0);

    for (int n0 = 0; n0 < KC; n0 += TROWS) {
        const int buf = (n0 / TROWS) & 1;
#if HAS_ASYNC_LDS
        __builtin_amdgcn_s_wait_asynccnt(0);   // my share of tile `buf` landed
#endif
        __syncthreads();                       // tile complete; prev compute done
        if (n0 + TROWS < KC) prefetch(buf ^ 1, n0 + TROWS);

        #pragma unroll
        for (int nt = 0; nt < TROWS / 16; ++nt) {
            // B fragment: lane = column n, K = [hl*8..+7] & [16+hl*8..+7] (matches A)
            const __bf16* brow = &lds[buf][(nt * 16 + l15) * LDSD + hl * 8];
            f32x8 acc0 = (f32x8){0.f,0.f,0.f,0.f,0.f,0.f,0.f,0.f};
            f32x8 acc1 = (f32x8){0.f,0.f,0.f,0.f,0.f,0.f,0.f,0.f};
            #pragma unroll
            for (int c = 0; c < 8; ++c) {
                bf16x8 lo = *(const bf16x8*)(brow + c * 32);
                bf16x8 hi = *(const bf16x8*)(brow + c * 32 + 16);
                bf16x16 bfrag = __builtin_shufflevector(lo, hi, 0,1,2,3,4,5,6,7,8,9,10,11,12,13,14,15);
                // one B fragment feeds two WMMAs (two M sets)
                acc0 = __builtin_amdgcn_wmma_f32_16x16x32_bf16(
                        false, afrag[0][c], false, bfrag, (short)0, acc0, false, false);
                acc1 = __builtin_amdgcn_wmma_f32_16x16x32_bf16(
                        false, afrag[1][c], false, bfrag, (short)0, acc1, false, false);
            }
            const int   n   = n0 + nt * 16 + l15;
            const float c2n = c2[n];
            #pragma unroll
            for (int r = 0; r < 8; ++r) {
                float d0 = x2row[0][r] + c2n - 2.0f * acc0[r];
                if (d0 < bestd[0][r]) { bestd[0][r] = d0; besti[0][r] = n; }
                float d1 = x2row[1][r] + c2n - 2.0f * acc1[r];
                if (d1 < bestd[1][r]) { bestd[1][r] = d1; besti[1][r] = n; }
            }
        }
    }

    // argmin reduce across the 16 column-lanes (xor 1,2,4,8 stays in half-wave)
    #pragma unroll
    for (int s = 0; s < 2; ++s) {
        #pragma unroll
        for (int r = 0; r < 8; ++r) {
            float bd = bestd[s][r]; int bi = besti[s][r];
            #pragma unroll
            for (int off = 1; off < 16; off <<= 1) {
                float od = __shfl_xor(bd, off, 32);
                int   oi = __shfl_xor(bi, off, 32);
                if (od < bd || (od == bd && oi < bi)) { bd = od; bi = oi; }  // first-match tie-break
            }
            if (l15 == 0) {
                const int row = m_base + s * 16 + hl * 8 + r;
                ind_ws[row]  = bi;
                ind_out[row] = (float)bi;
            }
        }
    }
}

// -------------------------------------------------------------------
// 3) zero accumulated outputs (loss, cluster_size, embed_sum)
// -------------------------------------------------------------------
__global__ void vq_zero(float* __restrict__ p, int n)
{
    for (int i = blockIdx.x * blockDim.x + threadIdx.x; i < n; i += gridDim.x * blockDim.x)
        p[i] = 0.f;
}

// -------------------------------------------------------------------
// 4) gather quantize (exact f32 codebook), commit loss, scatter stats
//    One block per point; thread = feature element.
// -------------------------------------------------------------------
__global__ void __launch_bounds__(256) vq_finalize(
    const float* __restrict__ x, const float* __restrict__ cb,
    const int* __restrict__ ind_ws,
    float* __restrict__ q_out, float* __restrict__ loss,
    float* __restrict__ cs, float* __restrict__ es)
{
    __shared__ float red[256];
    const int row = blockIdx.x;
    const int t   = threadIdx.x;
    const int ind = ind_ws[row];
    const float xv = x[(size_t)row * DIM + t];
    const float qv = cb[(size_t)ind * DIM + t];
    q_out[(size_t)row * DIM + t] = qv;          // straight-through fwd value == quantize
    const float diff = qv - xv;
    red[t] = diff * diff;
    atomicAdd(&es[(size_t)ind * DIM + t], xv);  // embed_sum scatter
    __syncthreads();
    for (int s = 128; s > 0; s >>= 1) { if (t < s) red[t] += red[t + s]; __syncthreads(); }
    if (t == 0) {
        atomicAdd(loss, red[0] * (1.0f / 8388608.0f));   // mean over BN*DIM
        atomicAdd(&cs[ind], 1.0f);
    }
}

extern "C" void kernel_launch(void* const* d_in, const int* in_sizes, int n_in,
                              void* d_out, int out_size, void* d_ws, size_t ws_size,
                              hipStream_t stream)
{
    const float* x  = (const float*)d_in[0];
    const float* cb = (const float*)d_in[1];

    char* ws = (char*)d_ws;
    __bf16* xb   = (__bf16*)(ws + WS_XB);
    __bf16* cbb  = (__bf16*)(ws + WS_CB);
    float*  x2   = (float*) (ws + WS_X2);
    float*  c2   = (float*) (ws + WS_C2);
    int*    indw = (int*)   (ws + WS_IND);

    float* out     = (float*)d_out;
    float* q_out   = out + OUT_Q;
    float* ind_out = out + OUT_IND;
    float* loss    = out + OUT_LOSS;
    float* cs      = out + OUT_CS;
    float* es      = out + OUT_ES;

    vq_prep    <<<BN + KC, 256, 0, stream>>>(x, cb, xb, cbb, x2, c2);
    vq_zero    <<<2048,    256, 0, stream>>>(loss, 1 + KC + KC * DIM);
    vq_argmin  <<<BN / 128,128, 0, stream>>>(xb, cbb, x2, c2, indw, ind_out);
    vq_finalize<<<BN,      256, 0, stream>>>(x, cb, indw, q_out, loss, cs, es);
}
